// UniversalConvModule_54365696032872
// MI455X (gfx1250) — matrix operand order repaired and checked by reading, not compiled, not run on hardware
//
#include <hip/hip_runtime.h>
#include <hip/hip_bf16.h>

// ---------------------------------------------------------------------------
// UniversalConvModule on MI455X (gfx1250):
//   Stage 1: per-sample MLP weight generator (tiny, VALU) -> d_ws as f16,
//            K padded 147 -> 160 with zeros so stage 2 needs no K bounds.
//   Stage 2: grouped conv as implicit GEMM with v_wmma_f32_16x16x32_f16.
//            M=64 out-ch, N=12544 pixels, K=147 (5 x K32 steps).
//   This rev: fix async-to-LDS builtin arg types (v4i AS1* / v4i AS3*).
// ---------------------------------------------------------------------------

typedef __attribute__((ext_vector_type(16))) _Float16 v16h;
typedef __attribute__((ext_vector_type(8)))  _Float16 v8h;
typedef __attribute__((ext_vector_type(8)))  float    v8f;

union V16H { v16h v; v8h h[2]; };

#define BATCH    32
#define CIN      3
#define HIN      224
#define WIN      224
#define OCH      64
#define OH       112
#define OW       112
#define NPIX     (OH * OW)       // 12544
#define KDIM     147             // 3*7*7
#define KPAD     160             // padded to 5 * 32
#define HIDDEN   128
#define ODIM     (OCH * KDIM)    // 9408

#if __has_builtin(__builtin_amdgcn_global_load_async_to_lds_b128)
#define HAVE_ASYNC_LDS 1
// Builtin signature (per hipcc diagnostic): (v4i AS1*, v4i AS3*, Imm int, Imm int)
typedef __attribute__((__vector_size__(16))) int vi4;
typedef __attribute__((address_space(1))) vi4 gvi4;
typedef __attribute__((address_space(3))) vi4 lvi4;
#else
#define HAVE_ASYNC_LDS 0
#endif

// ---------------------------------------------------------------------------
// Stage 1: weights[b] = relu(mask[b] @ W1 + b1) @ W2 + b2, stored f16,
// layout wsW[b][m][kpad] with kpad zero-padded 147..159.
// ---------------------------------------------------------------------------
__global__ __launch_bounds__(256)
void gen_weights_kernel(const float* __restrict__ mask,
                        const float* __restrict__ W1,
                        const float* __restrict__ b1,
                        const float* __restrict__ W2,
                        const float* __restrict__ b2,
                        _Float16* __restrict__ wsW) {
    __shared__ float h[HIDDEN];
    const int b   = blockIdx.x;
    const int tid = threadIdx.x;

    if (tid < HIDDEN) {
        float acc = b1[tid];
        #pragma unroll
        for (int i = 0; i < CIN; ++i)
            acc += mask[b * CIN + i] * W1[i * HIDDEN + tid];
        h[tid] = acc > 0.f ? acc : 0.f;
    }
    __syncthreads();

    for (int idx = tid; idx < OCH * KPAD; idx += blockDim.x) {
        const int m = idx / KPAD;
        const int k = idx - m * KPAD;
        float val = 0.f;
        if (k < KDIM) {
            const int o = m * KDIM + k;
            val = b2[o];
            for (int j = 0; j < HIDDEN; ++j)
                val += h[j] * W2[j * ODIM + o];
        }
        wsW[(size_t)b * (OCH * KPAD) + idx] = (_Float16)val;
    }
}

// ---------------------------------------------------------------------------
// Stage 2: per (batch, 64-pixel tile) block; 128 threads = 4 waves.
//   Wlds: 64 x 160 f16 weights (shared)   Plds[w]: 16 x 160 f16 im2col patch
//   Each wave: B-frag (patch) reused across 4 A tiles -> 4 accumulators
//   covering all 64 out-channels for its 16 pixels; 5 K32 steps = 20 WMMAs.
// ---------------------------------------------------------------------------
__global__ __launch_bounds__(128)
void dynconv_wmma_kernel(const float* __restrict__ x,
                         const _Float16* __restrict__ wsW,
                         float* __restrict__ out) {
    __shared__ alignas(16) _Float16 Wlds[OCH * KPAD];      // 20 KB
    __shared__ alignas(16) _Float16 Plds[4][16 * KPAD];    // 20 KB

    const int b    = blockIdx.y;
    const int tile = blockIdx.x;           // 0..195 (196 * 64 = 12544)
    const int tid  = threadIdx.x;
    const int wave = tid >> 5;
    const int lane = tid & 31;

    // ---- stage this sample's generated weights into LDS -------------------
    {
        const uint4* src = (const uint4*)(wsW + (size_t)b * (OCH * KPAD));
        uint4*       dst = (uint4*)Wlds;
#if HAVE_ASYNC_LDS
        // GLOBAL_LOAD_ASYNC_TO_LDS_B128: DMA 16B/lane, tracked by ASYNCcnt —
        // no VGPR round-trip, overlaps with the im2col gather below.
        for (int i = tid; i < (OCH * KPAD) / 8; i += 128)
            __builtin_amdgcn_global_load_async_to_lds_b128(
                (gvi4*)(src + i), (lvi4*)(dst + i), 0, 0);
#else
        for (int i = tid; i < (OCH * KPAD) / 8; i += 128)
            dst[i] = src[i];
#endif
    }

    // ---- im2col: each wave builds its 16 pixels x 160 K patch (f16) -------
    // 112 % 16 == 0, so a wave's 16 flat pixels share one output row:
    // oy / ox_base are wave constants. 160 = 5*32, so lane owns k-slots
    // k = lane + 32j, j=0..4; their (c,r,s,row,validity) hoist out of p-loop.
    const int pix_base = tile * 64 + wave * 16;
    const int oy   = pix_base / OW;            // wave-uniform
    const int oxb  = pix_base - oy * OW;
    const int iy0  = oy * 2 - 3;
    const int ixb  = oxb * 2 - 3;

    int          s_j[5];
    const float* row_j[5];
    bool         ok_j[5];
    #pragma unroll
    for (int j = 0; j < 5; ++j) {
        const int k  = lane + 32 * j;
        const int c  = k / 49;
        const int rs = k - c * 49;
        const int r  = rs / 7;
        s_j[j] = rs - r * 7;
        const int iy = iy0 + r;
        ok_j[j]  = (k < KDIM) & (iy >= 0) & (iy < HIN);
        row_j[j] = x + (((size_t)b * CIN + c) * HIN + iy) * (size_t)WIN;
    }

    for (int p = 0; p < 16; ++p) {
        const int ixp = ixb + p * 2;
        _Float16* dst = &Plds[wave][p * KPAD + lane];
        #pragma unroll
        for (int j = 0; j < 5; ++j) {
            const int ix = ixp + s_j[j];
            float v = 0.f;
            if (ok_j[j] && ix >= 0 && ix < WIN) v = row_j[j][ix];
            dst[32 * j] = (_Float16)v;
        }
    }

#if HAVE_ASYNC_LDS
#if __has_builtin(__builtin_amdgcn_s_wait_asynccnt)
    __builtin_amdgcn_s_wait_asynccnt(0);
#else
    asm volatile("s_wait_asynccnt 0x0" ::: "memory");
#endif
#endif
    __syncthreads();

    // ---- WMMA fragment gathers (ISA 7.12.2 layouts, 16B-aligned ds reads) --
    const int half = lane >> 4;     // lane group selects K sub-range
    const int mn   = lane & 15;     // A: row M; B: col N; C/D: col N

    v8f c0 = {}, c1 = {}, c2 = {}, c3 = {};

    #pragma unroll
    for (int kt = 0; kt < 5; ++kt) {
        const int kb = kt * 32;

        // B (patch) 32x16: lane n=mn holds elems 0..7 -> K = kb + half*8 + (0..7), elems 8..15 -> K = kb + 16 + half*8 + (0..7)
        V16H bf;
        bf.h[0] = *(const v8h*)&Plds[wave][mn * KPAD + kb + half * 8];
        bf.h[1] = *(const v8h*)&Plds[wave][mn * KPAD + kb + 16 + half * 8];

        // A (weights) 16x32 per M-tile: lane m=mn holds
        //   elems 0..7  -> K = kb + half*8 + (0..7)
        //   elems 8..15 -> K = kb + 16 + half*8 + (0..7)
        V16H a0, a1, a2, a3;
        a0.h[0] = *(const v8h*)&Wlds[( 0 + mn) * KPAD + kb + half * 8];
        a0.h[1] = *(const v8h*)&Wlds[( 0 + mn) * KPAD + kb + 16 + half * 8];
        a1.h[0] = *(const v8h*)&Wlds[(16 + mn) * KPAD + kb + half * 8];
        a1.h[1] = *(const v8h*)&Wlds[(16 + mn) * KPAD + kb + 16 + half * 8];
        a2.h[0] = *(const v8h*)&Wlds[(32 + mn) * KPAD + kb + half * 8];
        a2.h[1] = *(const v8h*)&Wlds[(32 + mn) * KPAD + kb + 16 + half * 8];
        a3.h[0] = *(const v8h*)&Wlds[(48 + mn) * KPAD + kb + half * 8];
        a3.h[1] = *(const v8h*)&Wlds[(48 + mn) * KPAD + kb + 16 + half * 8];

        c0 = __builtin_amdgcn_wmma_f32_16x16x32_f16(false, a0.v, false, bf.v,
                                                    (short)0, c0, false, false);
        c1 = __builtin_amdgcn_wmma_f32_16x16x32_f16(false, a1.v, false, bf.v,
                                                    (short)0, c1, false, false);
        c2 = __builtin_amdgcn_wmma_f32_16x16x32_f16(false, a2.v, false, bf.v,
                                                    (short)0, c2, false, false);
        c3 = __builtin_amdgcn_wmma_f32_16x16x32_f16(false, a3.v, false, bf.v,
                                                    (short)0, c3, false, false);
    }

    // ---- store: C/D elem r <-> M = r + half*8 ; lane mn <-> pixel n ----
    const int pg = pix_base + mn;
    float* outp = out + (size_t)b * OCH * NPIX + pg;
    #pragma unroll
    for (int r = 0; r < 8; ++r) {
        const int ml = half * 8 + r;
        outp[(size_t)( 0 + ml) * NPIX] = c0[r];
        outp[(size_t)(16 + ml) * NPIX] = c1[r];
        outp[(size_t)(32 + ml) * NPIX] = c2[r];
        outp[(size_t)(48 + ml) * NPIX] = c3[r];
    }
}

// ---------------------------------------------------------------------------
extern "C" void kernel_launch(void* const* d_in, const int* in_sizes, int n_in,
                              void* d_out, int out_size, void* d_ws, size_t ws_size,
                              hipStream_t stream) {
    const float* x    = (const float*)d_in[0];   // [32,3,224,224]
    const float* mask = (const float*)d_in[1];   // [32,3]
    const float* W1   = (const float*)d_in[2];   // [3,128]
    const float* b1   = (const float*)d_in[3];   // [128]
    const float* W2   = (const float*)d_in[4];   // [128,9408]
    const float* b2   = (const float*)d_in[5];   // [9408]
    float*       out  = (float*)d_out;           // [32,64,112,112]
    _Float16*    wsW  = (_Float16*)d_ws;         // [32,64,160] f16 (640 KB)

    gen_weights_kernel<<<dim3(BATCH), dim3(256), 0, stream>>>(mask, W1, b1, W2, b2, wsW);

    dim3 grid(NPIX / 64, BATCH);                 // 196 x 32 blocks
    dynconv_wmma_kernel<<<grid, dim3(128), 0, stream>>>(x, wsW, out);
}
